// HetSGG_56667798503484
// MI455X (gfx1250) — matrix operand order repaired and checked by reading, not compiled, not run on hardware
//
#include <hip/hip_runtime.h>
#include <hip/hip_bf16.h>
#include <math.h>

// ---------------------------------------------------------------------------
// HetSGG layer for MI455X (gfx1250, wave32, WMMA).
// Per-relation weights W_r precomputed from bases; edges bucketed by relation
// (buckets padded to 32); the 4 per-edge projections run as relation-uniform
// 32x128 @ 128x128 tiles with v_wmma_f32_16x16x32_bf16 (bf16 in, f32 acc).
// Each wave reuses every B fragment across two M-subtiles.
// ---------------------------------------------------------------------------

typedef __attribute__((ext_vector_type(16))) __bf16 bf16x16;
typedef __attribute__((ext_vector_type(8)))  float  f32x8;
typedef __attribute__((ext_vector_type(4)))  float  f32x4;
typedef __attribute__((ext_vector_type(4)))  int    i32x4;

#define CDIM 128
#define LDA  136   // LDS row stride in halves (272B) -> breaks 64-bank conflicts
#define TILE_M 32

__device__ __forceinline__ float leaky01(float x) { return x > 0.f ? x : 0.01f * x; }

// float atomic max via ordered-int trick (works with -inf init)
__device__ __forceinline__ void atomicMaxF(float* addr, float v) {
    if (v >= 0.f) atomicMax((int*)addr, __float_as_int(v));
    else          atomicMin((unsigned int*)addr, __float_as_uint(v));
}

// small[] integer scratch layout
#define S_CNT_REL 0
#define S_CNT_INV 16
#define S_CUR_REL 32
#define S_CUR_INV 48
#define S_OFF_REL 64
#define S_OFF_INV 80

// ---------------------------------------------------------------------------
__global__ void k_init(int* perm_rel, int* perm_inv, int PE, int* smallbuf,
                       float* segmax_s, float* segmax_o,
                       float* denom_s, float* denom_o, int NR,
                       float* agg_s, float* agg_o, int NC,
                       unsigned* mask_s, unsigned* mask_o, int N_) {
    int i = blockIdx.x * blockDim.x + threadIdx.x;
    if (i < 128) smallbuf[i] = 0;
    if (i < PE)  { perm_rel[i] = -1; perm_inv[i] = -1; }
    if (i < NR)  { segmax_s[i] = -INFINITY; segmax_o[i] = -INFINITY;
                   denom_s[i] = 0.f; denom_o[i] = 0.f; }
    if (i < NC)  { agg_s[i] = 0.f; agg_o[i] = 0.f; }
    if (i < N_)  { mask_s[i] = 0u; mask_o[i] = 0u; }
}

// W_r = sum_b att[r,b] * B_b, stored TRANSPOSED (Wt[r][d][c] = W_r[c][d]) in bf16
__global__ void k_weights(const float* b0, const float* a0,
                          const float* b1, const float* a1,
                          const float* b2, const float* a2,
                          const float* b3, const float* a3,
                          __bf16* wt, int R_, int NB_, int total) {
    int i = blockIdx.x * blockDim.x + threadIdx.x;
    if (i >= total) return;
    int cc   = i % CDIM;          // k (input channel)
    int tmp  = i / CDIM;
    int d    = tmp % CDIM;        // output channel -> row of Wt
    int tmp2 = tmp / CDIM;
    int r    = tmp2 % R_;
    int p    = tmp2 / R_;
    const float* basis = p == 0 ? b0 : p == 1 ? b1 : p == 2 ? b2 : b3;
    const float* att   = p == 0 ? a0 : p == 1 ? a1 : p == 2 ? a2 : a3;
    float s = 0.f;
    for (int b = 0; b < NB_; ++b)
        s += att[r * NB_ + b] * basis[((size_t)b * CDIM + cc) * CDIM + d];
    wt[((size_t)(p * R_ + r) * CDIM + d) * CDIM + cc] = (__bf16)s;
}

__global__ void k_count(const int* trel, const int* tinv, int* smallbuf, int E_) {
    int e = blockIdx.x * blockDim.x + threadIdx.x;
    if (e >= E_) return;
    atomicAdd(&smallbuf[S_CNT_REL + trel[e]], 1);
    atomicAdd(&smallbuf[S_CNT_INV + tinv[e]], 1);
}

__global__ void k_scan(int* smallbuf, int R_) {
    if (blockIdx.x == 0 && threadIdx.x == 0) {
        int o = 0;
        for (int r = 0; r < R_; ++r) {
            smallbuf[S_OFF_REL + r] = o; smallbuf[S_CUR_REL + r] = o;
            o += (smallbuf[S_CNT_REL + r] + (TILE_M - 1)) & ~(TILE_M - 1);
        }
        smallbuf[S_OFF_REL + R_] = o;
        o = 0;
        for (int r = 0; r < R_; ++r) {
            smallbuf[S_OFF_INV + r] = o; smallbuf[S_CUR_INV + r] = o;
            o += (smallbuf[S_CNT_INV + r] + (TILE_M - 1)) & ~(TILE_M - 1);
        }
        smallbuf[S_OFF_INV + R_] = o;
    }
}

__global__ void k_scatter(const int* trel, const int* tinv, int* smallbuf,
                          int* perm_rel, int* perm_inv, int E_) {
    int e = blockIdx.x * blockDim.x + threadIdx.x;
    if (e >= E_) return;
    int p = atomicAdd(&smallbuf[S_CUR_REL + trel[e]], 1); perm_rel[p] = e;
    int q = atomicAdd(&smallbuf[S_CUR_INV + tinv[e]], 1); perm_inv[q] = e;
}

// ---------------------------------------------------------------------------
// Relation-bucketed tile GEMM: 32 edges x 128 out-channels per workgroup.
// 8 wave32 waves; each wave computes two 16x16 D tiles sharing each B
// fragment: 8x v_wmma_f32_16x16x32_bf16 per wave (64 per workgroup) over K=128.
// rowmap==nullptr -> rows indexed directly by edge id (rel_embedding input).
// ---------------------------------------------------------------------------
__global__ __launch_bounds__(256) void k_gemm(
    const float* __restrict__ feat, const int* __restrict__ rowmap,
    const int* __restrict__ perm, const int* __restrict__ offs,
    const __bf16* __restrict__ wt, float* __restrict__ outmsg, int R_) {
    __shared__ int    s_off[17];
    __shared__ int    s_edge[TILE_M];
    __shared__ __bf16 As[TILE_M * LDA];

    int t = threadIdx.x;
    if (t <= R_) s_off[t] = offs[t];
    __syncthreads();

    int row0 = blockIdx.x * TILE_M;
    if (row0 >= s_off[R_]) return;          // block-uniform exit
    int r = 0;
    while (row0 >= s_off[r + 1]) ++r;       // bucket -> relation

    // cooperative gather: 32 rows x 128 f32 -> bf16 LDS tile (2 tasks/thread)
#pragma unroll
    for (int it = 0; it < 2; ++it) {
        int j    = t + it * 256;
        int m    = j >> 4;                  // row 0..31
        int cseg = (j & 15) << 3;           // 8-wide channel segment
        int e    = perm[row0 + m];
        if ((j & 15) == 0) s_edge[m] = e;
        union { __bf16 h[8]; i32x4 v; } u;
        if (e >= 0) {
            int grow = rowmap ? rowmap[e] : e;
            const float* srcp = feat + (size_t)grow * CDIM + cseg;
            f32x4 x0 = *(const f32x4*)(srcp);
            f32x4 x1 = *(const f32x4*)(srcp + 4);
            u.h[0] = (__bf16)x0[0]; u.h[1] = (__bf16)x0[1];
            u.h[2] = (__bf16)x0[2]; u.h[3] = (__bf16)x0[3];
            u.h[4] = (__bf16)x1[0]; u.h[5] = (__bf16)x1[1];
            u.h[6] = (__bf16)x1[2]; u.h[7] = (__bf16)x1[3];
        } else {
#pragma unroll
            for (int i2 = 0; i2 < 8; ++i2) u.h[i2] = (__bf16)0.0f;
        }
        *(i32x4*)&As[m * LDA + cseg] = u.v;
    }
    __syncthreads();

    int  wave = t >> 5, lane = t & 31;
    int  nloc = lane & 15;                  // = A row M (within subtile), = B col N
    bool hi   = lane >= 16;
    const __bf16* wcol = wt + ((size_t)r * CDIM + (wave << 4) + nloc) * CDIM;

    f32x8 acc0 = {}, acc1 = {};
#pragma unroll
    for (int kb = 0; kb < 4; ++kb) {
        bf16x16 a0, a1, b;
        // A fragment: lanes<16 hold K[0..7]+[16..23], lanes>=16 K[8..15]+[24..31]
        int ka0 = kb * 32 + (hi ? 8 : 0);
        int ka1 = kb * 32 + (hi ? 24 : 16);
        *(i32x4*)&a0         = *(const i32x4*)&As[nloc * LDA + ka0];
        *(((i32x4*)&a0) + 1) = *(const i32x4*)&As[nloc * LDA + ka1];
        *(i32x4*)&a1         = *(const i32x4*)&As[(16 + nloc) * LDA + ka0];
        *(((i32x4*)&a1) + 1) = *(const i32x4*)&As[(16 + nloc) * LDA + ka1];
        // B fragment: lanes<16 hold K[0..7]+[16..23] of col N, lanes>=16 K[8..15]+[24..31]
        int kbB = kb * 32 + (hi ? 8 : 0);
        *(i32x4*)&b         = *(const i32x4*)&wcol[kbB];
        *(((i32x4*)&b) + 1) = *(const i32x4*)&wcol[kbB + 16];
        acc0 = __builtin_amdgcn_wmma_f32_16x16x32_bf16(
            false, a0, false, b, (short)0, acc0, false, false);
        acc1 = __builtin_amdgcn_wmma_f32_16x16x32_bf16(
            false, a1, false, b, (short)0, acc1, false, false);
    }

    // preload the 8+8 edge ids this lane stores (C/D: lane<16 M=vv, else M=vv+8)
    int ebase = hi ? 8 : 0;
    i32x4 e0a = *(const i32x4*)&s_edge[ebase];
    i32x4 e0b = *(const i32x4*)&s_edge[ebase + 4];
    i32x4 e1a = *(const i32x4*)&s_edge[16 + ebase];
    i32x4 e1b = *(const i32x4*)&s_edge[16 + ebase + 4];

    int colo = (wave << 4) + nloc;
#pragma unroll
    for (int vv = 0; vv < 8; ++vv) {
        int ee0 = (vv < 4) ? e0a[vv & 3] : e0b[vv & 3];
        int ee1 = (vv < 4) ? e1a[vv & 3] : e1b[vv & 3];
        if (ee0 >= 0) outmsg[(size_t)ee0 * CDIM + colo] = acc0[vv];
        if (ee1 >= 0) outmsg[(size_t)ee1 * CDIM + colo] = acc1[vv];
    }
}

// ---------------------------------------------------------------------------
// rel_embedding = ef + s_msg*w0 + o_msg*w1, w = softmax(leaky(dot with w_ent))
__global__ __launch_bounds__(256) void k_relemb(
    const float* __restrict__ smsg, const float* __restrict__ omsg,
    const float* __restrict__ ef, const float* __restrict__ went,
    float* __restrict__ relout, int E_) {
    int e = blockIdx.x * 8 + (threadIdx.x >> 5);
    if (e >= E_) return;
    int lane = threadIdx.x & 31;
    const float* sp = smsg + (size_t)e * CDIM;
    const float* op = omsg + (size_t)e * CDIM;
    float ds = 0.f, dO = 0.f;
#pragma unroll
    for (int i = 0; i < 4; ++i) {
        int c = lane + 32 * i; float w = went[c];
        ds += sp[c] * w; dO += op[c] * w;
    }
#pragma unroll
    for (int off = 16; off > 0; off >>= 1) {
        ds += __shfl_xor(ds, off, 32);
        dO += __shfl_xor(dO, off, 32);
    }
    float a = leaky01(ds), b = leaky01(dO);
    float mx = fmaxf(a, b);
    float ea = __expf(a - mx), eb = __expf(b - mx);
    float inv = 1.f / (ea + eb);
    float w0 = ea * inv, w1 = eb * inv;
    float* rp = relout + (size_t)e * CDIM;
    const float* ep = ef + (size_t)e * CDIM;
#pragma unroll
    for (int i = 0; i < 4; ++i) {
        int c = lane + 32 * i;
        rp[c] = ep[c] + sp[c] * w0 + op[c] * w1;
    }
}

// attention logits + segment max + relation bitmask
__global__ __launch_bounds__(256) void k_escore(
    const float* __restrict__ msub, const float* __restrict__ mobj,
    const float* __restrict__ aw_s, const float* __restrict__ ab_s,
    const float* __restrict__ aw_o, const float* __restrict__ ab_o,
    const int* __restrict__ tinv, const int* __restrict__ trel,
    const int* __restrict__ src, const int* __restrict__ dst,
    float* esc_s, float* esc_o, float* segmax_s, float* segmax_o,
    unsigned* mask_s, unsigned* mask_o, int R_, int E_) {
    int e = blockIdx.x * 8 + (threadIdx.x >> 5);
    if (e >= E_) return;
    int lane = threadIdx.x & 31;
    int ts = tinv[e], to = trel[e];
    const float* ms = msub + (size_t)e * CDIM;
    const float* mo = mobj + (size_t)e * CDIM;
    float ds = 0.f, dO = 0.f;
#pragma unroll
    for (int i = 0; i < 4; ++i) {
        int c = lane + 32 * i;
        ds += ms[c] * aw_s[ts * CDIM + c];
        dO += mo[c] * aw_o[to * CDIM + c];
    }
#pragma unroll
    for (int off = 16; off > 0; off >>= 1) {
        ds += __shfl_xor(ds, off, 32);
        dO += __shfl_xor(dO, off, 32);
    }
    if (lane == 0) {
        float es = leaky01(ds + ab_s[ts]);
        float eo = leaky01(dO + ab_o[to]);
        esc_s[e] = es; esc_o[e] = eo;
        int is = src[e], io = dst[e];
        atomicMaxF(&segmax_s[is * R_ + ts], es);
        atomicMaxF(&segmax_o[io * R_ + to], eo);
        atomicOr(&mask_s[is], 1u << ts);
        atomicOr(&mask_o[io], 1u << to);
    }
}

__global__ void k_expsum(const int* tinv, const int* trel,
                         const int* src, const int* dst,
                         float* esc_s, float* esc_o,
                         const float* segmax_s, const float* segmax_o,
                         float* denom_s, float* denom_o, int R_, int E_) {
    int e = blockIdx.x * blockDim.x + threadIdx.x;
    if (e >= E_) return;
    int ss = src[e] * R_ + tinv[e];
    int so = dst[e] * R_ + trel[e];
    float xs = __expf(esc_s[e] - segmax_s[ss]);
    float xo = __expf(esc_o[e] - segmax_o[so]);
    esc_s[e] = xs; esc_o[e] = xo;
    atomicAdd(&denom_s[ss], xs);
    atomicAdd(&denom_o[so], xo);
}

__global__ __launch_bounds__(256) void k_agg(
    const float* __restrict__ msub, const float* __restrict__ mobj,
    const int* tinv, const int* trel, const int* src, const int* dst,
    const float* esc_s, const float* esc_o,
    const float* denom_s, const float* denom_o,
    float* agg_s, float* agg_o, int R_, int E_) {
    int e = blockIdx.x * 8 + (threadIdx.x >> 5);
    if (e >= E_) return;
    int lane = threadIdx.x & 31;
    int is = src[e], io = dst[e];
    float as = esc_s[e] / denom_s[is * R_ + tinv[e]];
    float ao = esc_o[e] / denom_o[io * R_ + trel[e]];
    const float* ms = msub + (size_t)e * CDIM;
    const float* mo = mobj + (size_t)e * CDIM;
#pragma unroll
    for (int i = 0; i < 4; ++i) {
        int c = lane + 32 * i;
        atomicAdd(&agg_s[(size_t)is * CDIM + c], as * ms[c]);
        atomicAdd(&agg_o[(size_t)io * CDIM + c], ao * mo[c]);
    }
}

__global__ void k_nodeout(const float* __restrict__ nf,
                          const float* __restrict__ agg_s,
                          const float* __restrict__ agg_o,
                          const unsigned* __restrict__ mask_s,
                          const unsigned* __restrict__ mask_o,
                          float* __restrict__ out, int N_) {
    int i = blockIdx.x * blockDim.x + threadIdx.x;
    if (i >= N_ * CDIM) return;
    int n = i >> 7;
    int ps = __popc(mask_s[n]); if (ps < 1) ps = 1;
    int po = __popc(mask_o[n]); if (po < 1) po = 1;
    out[i] = nf[i] + 0.5f * (agg_s[i] / (float)ps + agg_o[i] / (float)po);
}

// ---------------------------------------------------------------------------
extern "C" void kernel_launch(void* const* d_in, const int* in_sizes, int n_in,
                              void* d_out, int out_size, void* d_ws, size_t ws_size,
                              hipStream_t stream) {
    const int*   edge_index = (const int*)d_in[0];
    const float* nf     = (const float*)d_in[1];
    const float* ef     = (const float*)d_in[2];
    const int*   trel   = (const int*)d_in[3];
    const int*   tinv   = (const int*)d_in[4];
    const float* s2r_b  = (const float*)d_in[5];
    const float* s2r_a  = (const float*)d_in[6];
    const float* o2r_b  = (const float*)d_in[7];
    const float* o2r_a  = (const float*)d_in[8];
    const float* r2s_b  = (const float*)d_in[9];
    const float* r2s_a  = (const float*)d_in[10];
    const float* r2s_aw = (const float*)d_in[11];
    const float* r2s_ab = (const float*)d_in[12];
    const float* r2o_b  = (const float*)d_in[13];
    const float* r2o_a  = (const float*)d_in[14];
    const float* r2o_aw = (const float*)d_in[15];
    const float* r2o_ab = (const float*)d_in[16];
    const float* went   = (const float*)d_in[17];

    const int C  = in_sizes[17];               // 128
    const int E  = in_sizes[3];
    const int N  = in_sizes[1] / C;
    const int NB = in_sizes[5] / (C * C);
    const int R  = in_sizes[6] / NB;

    const int* src = edge_index;
    const int* dst = edge_index + E;

    const int PE = E + TILE_M * R;             // bucket-padded edge count

    // ---- workspace layout (256B aligned regions) ----
    char* wsb = (char*)d_ws;
    size_t off = 0;
    auto alloc = [&](size_t bytes) -> void* {
        void* p = wsb + off;
        off = (off + bytes + 255) & ~(size_t)255;
        return p;
    };
    __bf16* wt       = (__bf16*)alloc((size_t)4 * R * C * C * sizeof(__bf16));
    int* perm_rel    = (int*)alloc((size_t)PE * 4);
    int* perm_inv    = (int*)alloc((size_t)PE * 4);
    int* smallbuf    = (int*)alloc(128 * 4);
    float* s_msg     = (float*)alloc((size_t)E * C * 4);  // later reused: m_sub
    float* o_msg     = (float*)alloc((size_t)E * C * 4);  // later reused: m_obj
    float* esc_s     = (float*)alloc((size_t)E * 4);
    float* esc_o     = (float*)alloc((size_t)E * 4);
    float* segmax_s  = (float*)alloc((size_t)N * R * 4);
    float* segmax_o  = (float*)alloc((size_t)N * R * 4);
    float* denom_s   = (float*)alloc((size_t)N * R * 4);
    float* denom_o   = (float*)alloc((size_t)N * R * 4);
    float* agg_s     = (float*)alloc((size_t)N * C * 4);
    float* agg_o     = (float*)alloc((size_t)N * C * 4);
    unsigned* mask_s = (unsigned*)alloc((size_t)N * 4);
    unsigned* mask_o = (unsigned*)alloc((size_t)N * 4);
    (void)ws_size; (void)n_in;

    float* node_out = (float*)d_out;
    float* rel_out  = node_out + (size_t)N * C;
    (void)out_size;

    // ---- pipeline ----
    int initLen = N * C; if (PE > initLen) initLen = PE;
    if (N * R > initLen) initLen = N * R;
    k_init<<<(initLen + 255) / 256, 256, 0, stream>>>(
        perm_rel, perm_inv, PE, smallbuf,
        segmax_s, segmax_o, denom_s, denom_o, N * R,
        agg_s, agg_o, N * C, mask_s, mask_o, N);

    int wtot = 4 * R * C * C;
    k_weights<<<(wtot + 255) / 256, 256, 0, stream>>>(
        s2r_b, s2r_a, o2r_b, o2r_a, r2s_b, r2s_a, r2o_b, r2o_a,
        wt, R, NB, wtot);

    int egrid = (E + 255) / 256;
    k_count<<<egrid, 256, 0, stream>>>(trel, tinv, smallbuf, E);
    k_scan<<<1, 32, 0, stream>>>(smallbuf, R);
    k_scatter<<<egrid, 256, 0, stream>>>(trel, tinv, smallbuf, perm_rel, perm_inv, E);

    int ntiles = (PE + TILE_M - 1) / TILE_M;
    size_t wstride = (size_t)R * C * C;
    // entity -> relation projections
    k_gemm<<<ntiles, 256, 0, stream>>>(nf, src, perm_rel, smallbuf + S_OFF_REL,
                                       wt + 0 * wstride, s_msg, R);
    k_gemm<<<ntiles, 256, 0, stream>>>(nf, dst, perm_inv, smallbuf + S_OFF_INV,
                                       wt + 1 * wstride, o_msg, R);
    int ewgrid = (E + 7) / 8;
    k_relemb<<<ewgrid, 256, 0, stream>>>(s_msg, o_msg, ef, went, rel_out, E);

    // relation -> entity projections (reuse msg buffers as m_sub / m_obj)
    k_gemm<<<ntiles, 256, 0, stream>>>(rel_out, nullptr, perm_inv, smallbuf + S_OFF_INV,
                                       wt + 2 * wstride, s_msg, R);
    k_gemm<<<ntiles, 256, 0, stream>>>(rel_out, nullptr, perm_rel, smallbuf + S_OFF_REL,
                                       wt + 3 * wstride, o_msg, R);

    k_escore<<<ewgrid, 256, 0, stream>>>(s_msg, o_msg, r2s_aw, r2s_ab, r2o_aw, r2o_ab,
                                         tinv, trel, src, dst,
                                         esc_s, esc_o, segmax_s, segmax_o,
                                         mask_s, mask_o, R, E);
    k_expsum<<<egrid, 256, 0, stream>>>(tinv, trel, src, dst, esc_s, esc_o,
                                        segmax_s, segmax_o, denom_s, denom_o, R, E);
    k_agg<<<ewgrid, 256, 0, stream>>>(s_msg, o_msg, tinv, trel, src, dst,
                                      esc_s, esc_o, denom_s, denom_o,
                                      agg_s, agg_o, R, E);
    k_nodeout<<<(N * C + 255) / 256, 256, 0, stream>>>(nf, agg_s, agg_o,
                                                       mask_s, mask_o, node_out, N);
}
